// PSMMix3dEmbedding_59339268161764
// MI455X (gfx1250) — compile-verified, matrix operand order checked by reading, not run on hardware
//
#include <hip/hip_runtime.h>
#include <hip/hip_bf16.h>
#include <cmath>

#define BB 2
#define LL 384
#define DD 1024
#define KK 128
#define HH 32

typedef __attribute__((ext_vector_type(16))) _Float16 v16h;
typedef __attribute__((ext_vector_type(8)))  float    v8f;
typedef __attribute__((ext_vector_type(4)))  float    v4f;
typedef __attribute__((ext_vector_type(4)))  unsigned int v4u;
typedef __attribute__((ext_vector_type(8)))  int      v8i;
typedef __attribute__((ext_vector_type(4)))  int      v4i;

#if defined(__has_builtin)
#if __has_builtin(__builtin_amdgcn_tensor_load_to_lds) && __has_builtin(__builtin_amdgcn_s_wait_tensorcnt)
#define HAVE_TDM 1
#endif
#endif

union Frag16 {
    v16h v;
    _Float16 h[16];
    v4f f4[2];
};

__device__ __forceinline__ v8f vzero8() {
    v8f v;
#pragma unroll
    for (int i = 0; i < 8; i++) v[i] = 0.f;
    return v;
}

// ---------------------------------------------------------------------------
// Pack W1 (128x128) and W2 (32x128) fp32 -> f16 WMMA B-fragment order into one
// contiguous 40960-byte blob (frag1 | frag2) so the TDM can stage it in LDS.
// B element (k,n) = W[n][k]; lo lanes hold K {0..7,16..23} of each 32-chunk,
// hi lanes hold K {8..15,24..31}.  frag idx = ((nt*4+kc)*32 + lane)*16 + u
// ---------------------------------------------------------------------------
__global__ void k_pack(const float* __restrict__ W1, const float* __restrict__ W2,
                       _Float16* __restrict__ frag1, _Float16* __restrict__ frag2) {
    int idx = blockIdx.x * blockDim.x + threadIdx.x;
    if (idx >= 16384 + 4096) return;
    const float* W;
    _Float16* dst;
    int lidx;
    if (idx < 16384) { W = W1; dst = frag1; lidx = idx; }
    else             { W = W2; dst = frag2; lidx = idx - 16384; }
    int u    = lidx & 15;
    int lane = (lidx >> 4) & 31;
    int kc   = (lidx >> 9) & 3;
    int nt   = lidx >> 11;
    int hi   = lane >> 4;
    int n    = (lane & 15) + 16 * nt;
    int o    = (u & 7) + ((u & 8) << 1) + hi * 8;
    int k    = kc * 32 + o;
    dst[lidx] = (_Float16)W[n * KK + k];
}

// ---------------------------------------------------------------------------
// Fused: gaussian edge features -> GEMM1 (K=128,N=128, f16 WMMA) -> gelu ->
// GEMM2 (N=32) -> masked pos_attn_bias (B,H,L,L), plus ef = sum_j edge_feature.
// One wave per (b,i); j tiled by 16.  W1/W2 fragments staged in LDS via TDM.
// ---------------------------------------------------------------------------
__global__ __launch_bounds__(128) void k_edge(
    const int* __restrict__ token_id, const float* __restrict__ pos,
    const int* __restrict__ nte,
    const float* __restrict__ gmul, const float* __restrict__ gbias,
    const float* __restrict__ gmeans, const float* __restrict__ gstds,
    const _Float16* __restrict__ fragg,   // frag1|frag2 blob, 20480 halfs
    const float* __restrict__ b1, const float* __restrict__ b2,
    float* __restrict__ out_bias, float* __restrict__ ef_sum)
{
    __shared__ _Float16 s_frag[20480];            // 40960 B: W1 frags | W2 frags
    __shared__ float s_mean[KK], s_invstd[KK], s_coef[KK];
    __shared__ union {
        _Float16 gelu[4][16][136];                // [wave][row M][n], padded
        float    red[4][32][32];                  // ef reduction, 2 passes
    } sh;

    const int wave = threadIdx.x >> 5;
    const int lane = threadIdx.x & 31;
    const int hi   = lane >> 4;
    const int r    = lane & 15;
    const int gi   = blockIdx.x * 4 + wave;       // b*LL + i
    const int b    = gi / LL;
    const int i    = gi - b * LL;

    if (threadIdx.x < KK) {
        float st = fabsf(gstds[threadIdx.x]) + 1e-5f;
        s_mean[threadIdx.x]   = gmeans[threadIdx.x];
        s_invstd[threadIdx.x] = 1.f / st;
        s_coef[threadIdx.x]   = 0.3989422804014327f / st;  // 1/(sqrt(2pi)*std)
    }

#ifdef HAVE_TDM
    // Stage the 40 KB fragment blob into LDS with the Tensor Data Mover.
    if ((threadIdx.x >> 5) == 0) {                // wave 0 only (uniform branch)
        unsigned lds_base = (unsigned)(unsigned long long)&s_frag[0];
        unsigned long long ga = (unsigned long long)fragg;
        const int n8 = 5120;                      // 40960 B as 8-byte elements
        v4u g0;                                   // D# group 0
        g0[0] = 1u;                               // count=1, user mode
        g0[1] = lds_base;                         // lds_addr
        g0[2] = (unsigned)ga;                     // global_addr[31:0]
        g0[3] = (unsigned)((ga >> 32) & 0x1FFFFFFull) | (2u << 30);  // type=2
        v8i g1;                                   // D# group 1
        g1[0] = 3 << 16;                          // wg_mask=0, data_size=8B
        g1[1] = (n8 & 0xFFFF) << 16;              // tensor_dim0 lo16
        g1[2] = (n8 >> 16) | (1 << 16);           // tensor_dim0 hi16 | tensor_dim1=1
        g1[3] = n8 << 16;                         // tile_dim0
        g1[4] = 1;                                // tile_dim1=1, tile_dim2=0
        g1[5] = n8;                               // tensor_dim0_stride lo32
        g1[6] = (n8 & 0xFFFF) << 16;              // stride0 hi=0 | stride1 lo16
        g1[7] = n8 >> 16;                         // stride1 hi
        v4i g2, g3;
        g2[0] = 0; g2[1] = 0; g2[2] = 0; g2[3] = 0;
        g3[0] = 0; g3[1] = 0; g3[2] = 0; g3[3] = 0;
#if defined(__clang_major__) && (__clang_major__ >= 23)
        v8i g4;                                   // extra group (6-arg variant)
        g4[0] = 0; g4[1] = 0; g4[2] = 0; g4[3] = 0;
        g4[4] = 0; g4[5] = 0; g4[6] = 0; g4[7] = 0;
        __builtin_amdgcn_tensor_load_to_lds(g0, g1, g2, g3, g4, 0);
#else
        __builtin_amdgcn_tensor_load_to_lds(g0, g1, g2, g3, 0);
#endif
        __builtin_amdgcn_s_wait_tensorcnt(0);
    }
    __syncthreads();
    asm volatile("" ::: "memory");                // LDS written behind compiler's back
#else
    for (int q = threadIdx.x; q < 20480; q += 128) s_frag[q] = fragg[q];
    __syncthreads();
#endif

    const _Float16* lfrag1 = s_frag;              // 16384 halfs
    const _Float16* lfrag2 = s_frag + 16384;      // 4096 halfs

    const float pix = pos[(b * LL + i) * 3 + 0];
    const float piy = pos[(b * LL + i) * 3 + 1];
    const float piz = pos[(b * LL + i) * 3 + 2];
    const bool  pad_i = (token_id[b * LL + i] == 0);

    float ef_acc[64];
#pragma unroll
    for (int q = 0; q < 64; q++) ef_acc[q] = 0.f;

    for (int t = 0; t < LL / 16; t++) {
        const int j = t * 16 + r;                 // this lane's A-row pair
        float dx = pix - pos[(b * LL + j) * 3 + 0];
        float dy = piy - pos[(b * LL + j) * 3 + 1];
        float dz = piz - pos[(b * LL + j) * 3 + 2];
        float dist = sqrtf(fmaxf(dx * dx + dy * dy + dz * dz, 1e-12f));
        const long eoff = (((long)(b * LL + i)) * LL + j) * 2;
        int e0 = nte[eoff + 0], e1 = nte[eoff + 1];
        float mul = gmul[e0] + gmul[e1];
        float bia = gbias[e0] + gbias[e1];
        float y = mul * dist + bia;
        bool pad_j = (token_id[b * LL + j] == 0);

        // Build all 4 A1 fragments once (+ masked ef accumulation).
        Frag16 a1[4];
#pragma unroll
        for (int kc = 0; kc < 4; kc++) {
#pragma unroll
            for (int u = 0; u < 16; u++) {
                int o = (u & 7) + ((u & 8) << 1) + hi * 8;
                int k = kc * 32 + o;
                float aa = (y - s_mean[k]) * s_invstd[k];
                float g = __expf(-0.5f * aa * aa) * s_coef[k];
                a1[kc].h[u] = (_Float16)g;
                if (!pad_j) ef_acc[kc * 16 + u] += g;
            }
        }

        // GEMM1, nt-outer so only one v8f accumulator is live at a time.
#pragma unroll
        for (int nt = 0; nt < 8; nt++) {
            v8f acc = vzero8();
#pragma unroll
            for (int kc = 0; kc < 4; kc++) {
                Frag16 bf;
                const v4f* fp = (const v4f*)(lfrag1 + (((nt * 4 + kc) * 32 + lane) << 4));
                bf.f4[0] = fp[0];
                bf.f4[1] = fp[1];
                acc = __builtin_amdgcn_wmma_f32_16x16x32_f16(
                    false, a1[kc].v, false, bf.v, (short)0, acc, false, false);
            }
            // bias + exact gelu; re-layout D (row in VGPR) -> LDS for A2
            int n = r + nt * 16;
            float bb = b1[n];
#pragma unroll
            for (int rr = 0; rr < 8; rr++) {
                int M = rr + hi * 8;
                float v = acc[rr] + bb;
                float ge = 0.5f * v * (1.f + erff(v * 0.70710678118f));
                sh.gelu[wave][M][n] = (_Float16)ge;
            }
        }
        __syncthreads();

        v8f acc2[2];
        acc2[0] = vzero8();
        acc2[1] = vzero8();
#pragma unroll
        for (int kc = 0; kc < 4; kc++) {
            Frag16 a2;
            int k0 = kc * 32 + hi * 8;
#pragma unroll
            for (int u = 0; u < 8; u++) a2.h[u] = sh.gelu[wave][r][k0 + u];
#pragma unroll
            for (int u = 0; u < 8; u++) a2.h[8 + u] = sh.gelu[wave][r][k0 + 16 + u];
#pragma unroll
            for (int nt = 0; nt < 2; nt++) {
                Frag16 bf;
                const v4f* fp = (const v4f*)(lfrag2 + (((nt * 4 + kc) * 32 + lane) << 4));
                bf.f4[0] = fp[0];
                bf.f4[1] = fp[1];
                acc2[nt] = __builtin_amdgcn_wmma_f32_16x16x32_f16(
                    false, a2.v, false, bf.v, (short)0, acc2[nt], false, false);
            }
        }
        __syncthreads();

        // masked gab -> pos_attn_bias (B,H,L,L)
#pragma unroll
        for (int nt = 0; nt < 2; nt++) {
            int h = r + nt * 16;
            float bb = b2[h];
#pragma unroll
            for (int rr = 0; rr < 8; rr++) {
                int M = rr + hi * 8;
                int jj = t * 16 + M;
                float v = acc2[nt][rr] + bb;
                bool pj = (token_id[b * LL + jj] == 0);
                float ov = pad_i ? 0.f : (pj ? -__builtin_inff() : v);
                out_bias[(((long)(b * HH + h)) * LL + i) * LL + jj] = ov;
            }
        }
    }

    // ef reduction: two passes of 32 lane-private partials -> (b,i,K) sums
#pragma unroll
    for (int pass = 0; pass < 2; pass++) {
        __syncthreads();
#pragma unroll
        for (int q = 0; q < 32; q++) sh.red[wave][lane][q] = ef_acc[pass * 32 + q];
        __syncthreads();
        for (int k2 = lane; k2 < 64; k2 += 32) {
            int k = pass * 64 + k2;
            int o = k & 31;
            int kc = (k >> 5) & 1;               // chunk index within this pass
            int base = (o & 8) ? 16 : 0;         // which lane-half holds k
            int u = (o & 7) + ((o & 16) ? 8 : 0);
            float s = 0.f;
#pragma unroll
            for (int s2 = 0; s2 < 16; s2++) s += sh.red[wave][base + s2][kc * 16 + u];
            ef_sum[(b * LL + i) * KK + k] = s;
        }
    }
}

// ---------------------------------------------------------------------------
// d = gab.sum(heads), masks, softmax over j -> attn (B,L,L)
// ---------------------------------------------------------------------------
#define MINV (-3.4028234663852886e+38f)
__global__ __launch_bounds__(384) void k_softmax(
    const float* __restrict__ out_bias, const int* __restrict__ token_id,
    const unsigned char* __restrict__ adj, const unsigned char* __restrict__ is_periodic,
    float* __restrict__ attn)
{
    __shared__ float sbuf[384];
    int gi = blockIdx.x;
    int b = gi / LL, i = gi - b * LL;
    int j = threadIdx.x;
    float d = 0.f;
    for (int h = 0; h < HH; h++)
        d += out_bias[(((long)(b * HH + h)) * LL + i) * LL + j];
    bool pad_i = token_id[b * LL + i] == 0;
    bool pad_j = token_id[b * LL + j] == 0;
    bool molecule = (token_id[b * LL + i] <= 129) && (is_periodic[b] == 0);
    bool adj_eff = (adj[((long)(b * LL + i)) * LL + j] != 0) || (!molecule);
    if (pad_j || pad_i || !adj_eff) d = MINV;
    d *= 0.08838834764831845f;                       // K^-0.5

    sbuf[j] = d;
    __syncthreads();
    if (j < 128) sbuf[j] = fmaxf(sbuf[j], fmaxf(sbuf[j + 128], sbuf[j + 256]));
    __syncthreads();
    for (int s = 64; s > 0; s >>= 1) {
        if (j < s) sbuf[j] = fmaxf(sbuf[j], sbuf[j + s]);
        __syncthreads();
    }
    float mx = sbuf[0];
    __syncthreads();
    float e = __expf(d - mx);
    sbuf[j] = e;
    __syncthreads();
    if (j < 128) sbuf[j] = sbuf[j] + sbuf[j + 128] + sbuf[j + 256];
    __syncthreads();
    for (int s = 64; s > 0; s >>= 1) {
        if (j < s) sbuf[j] += sbuf[j + s];
        __syncthreads();
    }
    attn[((long)gi) * LL + j] = e / sbuf[0];
}

// pos_e = mask(pos @ Wpos^T)
__global__ void k_pose(const float* __restrict__ pos, const float* __restrict__ Wpos,
                       const int* __restrict__ token_id, float* __restrict__ pos_e) {
    int idx = blockIdx.x * blockDim.x + threadIdx.x;
    if (idx >= BB * LL * KK) return;
    int k = idx & (KK - 1);
    int row = idx >> 7;                               // b*LL + j
    float v = pos[row * 3 + 0] * Wpos[k * 3 + 0]
            + pos[row * 3 + 1] * Wpos[k * 3 + 1]
            + pos[row * 3 + 2] * Wpos[k * 3 + 2];
    pos_e[idx] = (token_id[row] == 0) ? 0.f : v;
}

// pfe = attn @ pos_e
__global__ __launch_bounds__(128) void k_pfe(const float* __restrict__ attn,
                                             const float* __restrict__ pos_e,
                                             float* __restrict__ pfe) {
    int gi = blockIdx.x;                              // b*LL + i
    int b = gi / LL;
    int k = threadIdx.x;
    float acc = 0.f;
    for (int j = 0; j < LL; j++)
        acc += attn[(long)gi * LL + j] * pos_e[(b * LL + j) * KK + k];
    pfe[gi * KK + k] = acc;
}

// pos_embedding = mask(pfe @ Wpfe^T + ef @ Wproj^T + bproj)
__global__ __launch_bounds__(256) void k_posembed(
    const float* __restrict__ pfe, const float* __restrict__ ef_sum,
    const float* __restrict__ Wpfe, const float* __restrict__ Wproj,
    const float* __restrict__ bproj, const int* __restrict__ token_id,
    float* __restrict__ out_pe) {
    __shared__ float s_pfe[KK], s_ef[KK];
    int gi = blockIdx.x;
    if (threadIdx.x < KK) {
        s_pfe[threadIdx.x] = pfe[gi * KK + threadIdx.x];
        s_ef[threadIdx.x] = ef_sum[gi * KK + threadIdx.x];
    }
    __syncthreads();
    bool pad_i = token_id[gi] == 0;
    for (int d = threadIdx.x; d < DD; d += 256) {
        float acc = bproj[d];
        for (int k = 0; k < KK; k++)
            acc += s_pfe[k] * Wpfe[d * KK + k] + s_ef[k] * Wproj[d * KK + k];
        out_pe[(long)gi * DD + d] = pad_i ? 0.f : acc;
    }
}

// sinusoidal time encoding -> silu(Wt1) hidden
__global__ __launch_bounds__(256) void k_time1(
    const float* __restrict__ time_step, const unsigned char* __restrict__ clean_mask,
    const float* __restrict__ Wt1, const float* __restrict__ bt1,
    float* __restrict__ hidden) {
    __shared__ float s_emb[DD];
    int gi = blockIdx.x;
    float t = (clean_mask[gi] ? 0.f : time_step[gi]) * 1000.f;
    for (int k = threadIdx.x; k < DD; k += 256) {
        int kk = (k < DD / 2) ? k : (k - DD / 2);
        float f = __expf(-9.210340371976184f * (float)kk / (float)(DD / 2));
        float ang = t * f;
        s_emb[k] = (k < DD / 2) ? __sinf(ang) : __cosf(ang);
    }
    __syncthreads();
    for (int d = threadIdx.x; d < DD; d += 256) {
        float acc = bt1[d];
        for (int k = 0; k < DD; k++) acc += s_emb[k] * Wt1[(long)d * DD + k];
        hidden[(long)gi * DD + d] = acc / (1.f + __expf(-acc));   // silu
    }
}

// time_embed = hidden @ Wt2^T + bt2
__global__ __launch_bounds__(256) void k_time2(
    const float* __restrict__ hidden, const float* __restrict__ Wt2,
    const float* __restrict__ bt2, float* __restrict__ out_time) {
    __shared__ float s_h[DD];
    int gi = blockIdx.x;
    for (int k = threadIdx.x; k < DD; k += 256) s_h[k] = hidden[(long)gi * DD + k];
    __syncthreads();
    for (int d = threadIdx.x; d < DD; d += 256) {
        float acc = bt2[d];
        for (int k = 0; k < DD; k++) acc += s_h[k] * Wt2[(long)d * DD + k];
        out_time[(long)gi * DD + d] = acc;
    }
}

// x = embed_w[token] + time_embed + pos_embedding; also emit padding_mask
__global__ void k_final(const int* __restrict__ token_id, const float* __restrict__ embed_w,
                        const float* __restrict__ out_time, const float* __restrict__ out_pe,
                        float* __restrict__ out_x, float* __restrict__ out_mask) {
    int idx = blockIdx.x * blockDim.x + threadIdx.x;
    if (idx >= BB * LL * DD) return;
    int row = idx / DD;
    int d = idx - row * DD;
    out_x[idx] = embed_w[token_id[row] * DD + d] + out_time[idx] + out_pe[idx];
    if (idx < BB * LL) out_mask[idx] = (token_id[idx] == 0) ? 1.f : 0.f;
}

extern "C" void kernel_launch(void* const* d_in, const int* in_sizes, int n_in,
                              void* d_out, int out_size, void* d_ws, size_t ws_size,
                              hipStream_t stream) {
    const int* token_id            = (const int*)d_in[0];
    const unsigned char* is_per    = (const unsigned char*)d_in[1];
    const float* pos               = (const float*)d_in[2];
    const unsigned char* adj       = (const unsigned char*)d_in[3];
    const int* nte                 = (const int*)d_in[4];
    const float* time_step         = (const float*)d_in[5];
    const unsigned char* clean     = (const unsigned char*)d_in[6];
    const float* embed_w           = (const float*)d_in[7];
    const float* Wpos              = (const float*)d_in[8];
    const float* Wpfe              = (const float*)d_in[9];
    const float* gmeans            = (const float*)d_in[10];
    const float* gstds             = (const float*)d_in[11];
    const float* gmul              = (const float*)d_in[12];
    const float* gbias             = (const float*)d_in[13];
    const float* W1                = (const float*)d_in[14];
    const float* b1                = (const float*)d_in[15];
    const float* W2                = (const float*)d_in[16];
    const float* b2                = (const float*)d_in[17];
    const float* Wproj             = (const float*)d_in[18];
    const float* bproj             = (const float*)d_in[19];
    const float* Wt1               = (const float*)d_in[20];
    const float* bt1               = (const float*)d_in[21];
    const float* Wt2               = (const float*)d_in[22];
    const float* bt2               = (const float*)d_in[23];
    (void)in_sizes; (void)n_in; (void)out_size; (void)ws_size;

    char* ws = (char*)d_ws;
    _Float16* frag1 = (_Float16*)(ws + 0);          // 32768 B  (blob start)
    _Float16* frag2 = (_Float16*)(ws + 32768);      // 8192 B   (blob tail)
    float* ef_sum   = (float*)(ws + 40960);         // 393216 B
    float* attn     = (float*)(ws + 434176);        // 1179648 B
    float* pos_e    = (float*)(ws + 1613824);       // 393216 B
    float* pfe      = (float*)(ws + 2007040);       // 393216 B
    float* hidden   = (float*)(ws + 2400256);       // 3145728 B

    float* out      = (float*)d_out;
    float* out_x    = out;                  // (B,L,D)
    float* out_mask = out + 786432;         // (B,L)
    float* out_time = out + 787200;         // (B,L,D)
    float* out_bias = out + 1573632;        // (B,H,L,L)
    float* out_pe   = out + 11010816;       // (B,L,D)

    k_pack<<<(20480 + 255) / 256, 256, 0, stream>>>(W1, W2, frag1, frag2);
    k_edge<<<(BB * LL) / 4, 128, 0, stream>>>(token_id, pos, nte, gmul, gbias,
                                              gmeans, gstds, frag1,
                                              b1, b2, out_bias, ef_sum);
    k_softmax<<<BB * LL, 384, 0, stream>>>(out_bias, token_id, adj, is_per, attn);
    k_pose<<<(BB * LL * KK + 255) / 256, 256, 0, stream>>>(pos, Wpos, token_id, pos_e);
    k_pfe<<<BB * LL, 128, 0, stream>>>(attn, pos_e, pfe);
    k_posembed<<<BB * LL, 256, 0, stream>>>(pfe, ef_sum, Wpfe, Wproj, bproj, token_id, out_pe);
    k_time1<<<BB * LL, 256, 0, stream>>>(time_step, clean, Wt1, bt1, hidden);
    k_time2<<<BB * LL, 256, 0, stream>>>(hidden, Wt2, bt2, out_time);
    k_final<<<(BB * LL * DD + 255) / 256, 256, 0, stream>>>(token_id, embed_w,
                                                            out_time, out_pe,
                                                            out_x, out_mask);
}
